// SpGAT_71236327571611
// MI455X (gfx1250) — matrix operand order repaired and checked by reading, not compiled, not run on hardware
//
#include <hip/hip_runtime.h>
#include <cstdint>
#include <cstddef>

// ---------------- problem constants (match reference) ----------------
#define NN    50000      // nodes
#define NR    500        // relations
#define E1C   200000     // direct edges
#define E2C   50000      // 2-hop edges
#define ETOT  (E1C + E2C)
#define DIN   128
#define DCAT  256        // NHEADS*D_OUT (layer1) == D_OUT (layer2)
#define ALPHA_LRELU 0.2f

// ---------------- vector types (trivial, union-safe) ----------------
typedef __attribute__((ext_vector_type(16))) __bf16 v16bf;
typedef __attribute__((ext_vector_type(8)))  float  v8f;
typedef __attribute__((ext_vector_type(4)))  unsigned int u32x4;
typedef __attribute__((ext_vector_type(4)))  float  f32x4;

union FragU { u32x4 u[2]; v16bf v; };   // 32B: one wmma A or B operand per lane
union FeatU { f32x4 q[2]; float f[8]; };

__device__ __forceinline__ unsigned short f2bf(float f) {
    union { float f; unsigned int u; } x; x.f = f;
    unsigned int u = x.u;
    return (unsigned short)((u + 0x7FFFu + ((u >> 16) & 1u)) >> 16);  // RNE
}

// ------------------------------------------------------------------
// Pre-convert weight matrices to bf16 in WMMA B-fragment order.
// B layout (16x16x32 bf16): lane l holds column n = nt*16 + (l&15);
// elements j=0..15 hold K = ks*32 + 16*(l>>4) + j (contiguous k order chosen here, same as A).
// bf1: layer1 B' [384 x 256], B'[k][h*128+o] = a_heads[h][o][k]
// bf2: layer2 B' [768 x 256], B'[k][o]       = a_out[o][k]
// ------------------------------------------------------------------
__global__ void build_bfrags(const float* __restrict__ a_heads,
                             const float* __restrict__ a_out,
                             unsigned short* __restrict__ bf1,
                             unsigned short* __restrict__ bf2) {
    int idx = blockIdx.x * blockDim.x + threadIdx.x;
    const int L1 = 16 * 12 * 512;             // nt*ks*lane*j
    const int L2 = 16 * 24 * 512;
    if (idx < L1) {
        int j = idx & 15, lane = (idx >> 4) & 31;
        int ks = (idx >> 9) % 12, nt = idx / (12 * 512);
        int n = nt * 16 + (lane & 15);
        int k = ks * 32 + ((lane >> 4) << 4) + j;
        int head = n >> 7, o = n & 127;
        bf1[idx] = f2bf(a_heads[(size_t)(head * 128 + o) * 384 + k]);
    } else if (idx < L1 + L2) {
        int t = idx - L1;
        int j = t & 15, lane = (t >> 4) & 31;
        int ks = (t >> 9) % 24, nt = t / (24 * 512);
        int n = nt * 16 + (lane & 15);
        int k = ks * 32 + ((lane >> 4) << 4) + j;
        bf2[t] = f2bf(a_out[(size_t)n * 768 + k]);
    }
}

// ---------------- out_rel = rel @ W   [500x128]x[128x256], fp32 ----------------
__global__ void outrel_kernel(const float* __restrict__ rel,
                              const float* __restrict__ W,
                              float* __restrict__ orel) {
    int idx = blockIdx.x * blockDim.x + threadIdx.x;
    if (idx >= NR * DCAT) return;
    int r = idx / DCAT, o = idx % DCAT;
    float acc = 0.f;
    #pragma unroll 8
    for (int k = 0; k < DIN; ++k) acc += rel[r * DIN + k] * W[k * DCAT + o];
    orel[idx] = acc;
}

// ------------------------------------------------------------------
// Fused edge kernel: gather -> bf16 WMMA edge transform -> attention
// logit (shfl_xor half-wave reduce) -> exp(-leaky_relu) -> fused
// segment-sum via global f32 atomics into H and rowsum.
// Workgroup = 512 threads = 16 wave32s; wave nt owns output cols
// [nt*16, nt*16+16); each block iteration = 32 edges (two 16-row
// tiles, two independent accumulators per wave -> 2x WMMA per epoch,
// B fragment reused for both, software-pipelined LDS A feeds).
// LAYER==1: x=ent[N,128], relv=rel[500,128], K=384, 2 heads.
// LAYER==2: x=out1[N,256], relv=out_rel[500,256], K=768, 1 head.
// ------------------------------------------------------------------
template <int LAYER>
__global__ __launch_bounds__(512) void edge_kernel(
    const float* __restrict__ x, const float* __restrict__ relv,
    const int* __restrict__ el, const int* __restrict__ rt,
    const int* __restrict__ eln, const int* __restrict__ rtn,
    const unsigned short* __restrict__ bfrag,
    const float* __restrict__ a2,
    float* __restrict__ H, float* __restrict__ rowsum)
{
    constexpr int XD = (LAYER == 1) ? 128 : 256;   // feature width per segment
    constexpr int K  = 3 * XD;                     // src | dst | edge-emb
    constexpr int KS = K / 32;                     // wmma K steps
    constexpr int NH = (LAYER == 1) ? 2 : 1;
    constexpr int CHF = XD / 32;                   // floats per lane per segment
    constexpr int CHQ = CHF / 4;                   // float4s per lane per segment

    __shared__ __align__(16) unsigned short sfeat[32][K];  // 32 edges x K bf16
    __shared__ float lds_lp[16][32];   // [ntile][edge] logit partials
    __shared__ float ee_lds[NH][32];   // attention weights per head per edge
    __shared__ int   snode[32];        // src node per edge (segment id)

    const int tid  = threadIdx.x;
    const int lane = tid & 31;
    const int nt   = tid >> 5;         // wave id == output column tile
    const int m16  = lane & 15;
    const int h2   = lane >> 4;
    const int ntiles = (ETOT + 31) / 32;   // 7813 (last pair is half)

    const float a2v   = a2[nt * 16 + m16];          // a2 col for this lane
    const int   headW = (LAYER == 1) ? (nt >> 3) : 0;

    for (int tile = blockIdx.x; tile < ntiles; tile += gridDim.x) {
        const int  ebase   = tile * 32;
        const bool t1valid = (ebase + 16) < ETOT;   // block-uniform

        // ---------------- stage up to 32 edges' features into LDS (bf16) ----------------
        #pragma unroll
        for (int t2 = 0; t2 < 2; ++t2) {
            if (t2 == 1 && !t1valid) break;         // uniform branch
            const int e_loc = nt + 16 * t2;         // wave nt stages rows nt, nt+16
            const int e = ebase + e_loc;
            int s, d;
            if (e < E1C) { s = el[e];        d = el[E1C + e]; }
            else         { int en = e - E1C; s = eln[en]; d = eln[E2C + en]; }
            if (lane == 0) {
                snode[e_loc] = s;
                int enext = (tile + gridDim.x) * 32 + e_loc;  // prefetch next tile idx
                if (enext < ETOT) {
                    if (enext < E1C) __builtin_prefetch(&el[enext], 0, 0);
                    else             __builtin_prefetch(&eln[enext - E1C], 0, 0);
                }
            }
            const int off = lane * CHF;
            FeatU S, D, Ev;
            const f32x4* sp = (const f32x4*)&x[(size_t)s * XD + off];
            const f32x4* dp = (const f32x4*)&x[(size_t)d * XD + off];
            #pragma unroll
            for (int i = 0; i < CHQ; ++i) { S.q[i] = sp[i]; D.q[i] = dp[i]; }
            if (e < E1C) {
                const f32x4* rp = (const f32x4*)&relv[(size_t)rt[e] * XD + off];
                #pragma unroll
                for (int i = 0; i < CHQ; ++i) Ev.q[i] = rp[i];
                if (LAYER == 1) {              // layer1 direct: rel + src + dst
                    #pragma unroll
                    for (int i = 0; i < CHF; ++i) Ev.f[i] += S.f[i] + D.f[i];
                }                               // layer2 direct: out_rel only
            } else {
                int en = e - E1C;
                int r0 = rtn[2 * en], r1 = rtn[2 * en + 1];
                const f32x4* rp0 = (const f32x4*)&relv[(size_t)r0 * XD + off];
                const f32x4* rp1 = (const f32x4*)&relv[(size_t)r1 * XD + off];
                FeatU R0, R1;
                #pragma unroll
                for (int i = 0; i < CHQ; ++i) { R0.q[i] = rp0[i]; R1.q[i] = rp1[i]; }
                #pragma unroll
                for (int i = 0; i < CHF; ++i)
                    Ev.f[i] = R0.f[i] + R1.f[i] + S.f[i] + D.f[i];
            }
            #pragma unroll
            for (int i = 0; i < CHF; ++i) {
                sfeat[e_loc][         off + i] = f2bf(S.f[i]);
                sfeat[e_loc][XD     + off + i] = f2bf(D.f[i]);
                sfeat[e_loc][2 * XD + off + i] = f2bf(Ev.f[i]);
            }
        }
        __syncthreads();

        // ---------------- edge_m tiles: 2x [16 x 16] via bf16 WMMA, pipelined ----------------
        // A (16x32): lane = row; elems 0..15 = K[16h2..16h2+15]
        auto loadA = [&](int ks, int row) {
            FragU A;
            A.u[0] = *(const u32x4*)&sfeat[row][ks * 32 + 16 * h2];
            A.u[1] = *(const u32x4*)&sfeat[row][ks * 32 + 16 * h2 + 8];
            return A;
        };
        auto loadB = [&](int ks) {
            FragU B;
            const u32x4* bp = (const u32x4*)&bfrag[((size_t)(nt * KS + ks) * 32 + lane) * 16];
            B.u[0] = bp[0]; B.u[1] = bp[1];
            return B;
        };
        v8f acc0 = {}, acc1 = {};
        FragU a0 = loadA(0, m16), a1 = loadA(0, 16 + m16), b = loadB(0);
        #pragma unroll
        for (int ks = 0; ks < KS; ++ks) {
            FragU a0n = a0, a1n = a1, bn = b;
            if (ks + 1 < KS) {                       // issue next loads before WMMAs
                a0n = loadA(ks + 1, m16);
                a1n = loadA(ks + 1, 16 + m16);
                bn  = loadB(ks + 1);
            }
            acc0 = __builtin_amdgcn_wmma_f32_16x16x32_bf16(
                false, a0.v, false, b.v, (short)0, acc0, false, false);
            acc1 = __builtin_amdgcn_wmma_f32_16x16x32_bf16(
                false, a1.v, false, b.v, (short)0, acc1, false, false);
            a0 = a0n; a1 = a1n; b = bn;
        }

        // ---------------- logit partials: sum_n a2[n]*edge_m[e][n] over our 16 cols ----------------
        float p0[8], p1[8];
        #pragma unroll
        for (int r = 0; r < 8; ++r) { p0[r] = a2v * acc0[r]; p1[r] = a2v * acc1[r]; }
        #pragma unroll
        for (int r = 0; r < 8; ++r) {
            for (int msk = 1; msk < 16; msk <<= 1) {
                p0[r] += __shfl_xor(p0[r], msk, 32);  // reduce within half-wave
                p1[r] += __shfl_xor(p1[r], msk, 32);
            }
        }
        if (m16 == 0) {
            #pragma unroll
            for (int r = 0; r < 8; ++r) {
                lds_lp[nt][     8 * h2 + r] = p0[r];
                lds_lp[nt][16 + 8 * h2 + r] = p1[r];
            }
        }
        __syncthreads();

        // ---------------- edge_e = exp(-leaky_relu(logit)) ; rowsum atomics ----------------
        if (tid < 32 * NH) {
            int head = tid >> 5, mm = tid & 31;
            if (mm < 16 || t1valid) {
                float lg = 0.f;
                #pragma unroll
                for (int i = 0; i < 16 / NH; ++i) lg += lds_lp[head * (16 / NH) + i][mm];
                float lr = lg > 0.f ? lg : ALPHA_LRELU * lg;
                float ee = expf(-lr);
                ee_lds[head][mm] = ee;
                atomicAdd(&rowsum[(size_t)head * NN + snode[mm]], ee);
            }
        }
        __syncthreads();

        // ---------------- fused segment-sum: H[src] += ee * edge_m (L2 atomics) ----------------
        #pragma unroll
        for (int r = 0; r < 8; ++r) {                // C layout: row = r + 8*(lane/16)
            int mm = 8 * h2 + r;
            float ee = ee_lds[headW][mm];
            atomicAdd(&H[(size_t)snode[mm] * DCAT + nt * 16 + m16], ee * acc0[r]);
        }
        if (t1valid) {                               // uniform branch
            #pragma unroll
            for (int r = 0; r < 8; ++r) {
                int mm = 16 + 8 * h2 + r;
                float ee = ee_lds[headW][mm];
                atomicAdd(&H[(size_t)snode[mm] * DCAT + nt * 16 + m16], ee * acc1[r]);
            }
        }
        __syncthreads();   // before next tile's staging overwrites sfeat
    }
}

// ---------------- h = elu(H / rowsum) in place ----------------
__global__ void normalize_kernel(float* __restrict__ H,
                                 const float* __restrict__ rowsum, int nh) {
    size_t idx = (size_t)blockIdx.x * blockDim.x + threadIdx.x;
    if (idx >= (size_t)NN * DCAT) return;
    int c = (int)(idx & 255);
    size_t n = idx >> 8;
    int head = (c * nh) >> 8;
    float rs = rowsum[(size_t)head * NN + n];
    rs = (rs == 0.f) ? 1e-12f : rs;
    float v = H[idx] / rs;
    H[idx] = v > 0.f ? v : (expf(v) - 1.f);
}

// ------------------------------------------------------------------
extern "C" void kernel_launch(void* const* d_in, const int* in_sizes, int n_in,
                              void* d_out, int out_size, void* d_ws, size_t ws_size,
                              hipStream_t stream) {
    (void)in_sizes; (void)n_in; (void)out_size; (void)ws_size;
    const float* ent      = (const float*)d_in[0];   // [NN,128]
    const float* rel      = (const float*)d_in[1];   // [NR,128]
    const int*   el       = (const int*)  d_in[2];   // [2,E1]
    const int*   rt       = (const int*)  d_in[3];   // [E1]
    const int*   eln      = (const int*)  d_in[4];   // [2,E2]
    const int*   rtn      = (const int*)  d_in[5];   // [E2,2]
    const float* a_heads  = (const float*)d_in[6];   // [2,128,384]
    const float* a2_heads = (const float*)d_in[7];   // [2,128] -> flat == col index
    const float* W        = (const float*)d_in[8];   // [128,256]
    const float* a_out    = (const float*)d_in[9];   // [256,768]
    const float* a2_out   = (const float*)d_in[10];  // [256]

    float* out_ent = (float*)d_out;                       // [NN,256]: H2 accum -> out
    float* orel    = (float*)d_out + (size_t)NN * DCAT;   // [NR,256]

    char*  ws  = (char*)d_ws;
    float* H1  = (float*)ws;                                    // NN*256 f32
    size_t offRs = (size_t)NN * DCAT * 4;
    float* rs1 = (float*)(ws + offRs);                          // 2*NN
    float* rs2 = rs1 + 2 * NN;                                  // NN
    size_t offBf = (offRs + (size_t)3 * NN * 4 + 255) & ~(size_t)255;
    unsigned short* bf1 = (unsigned short*)(ws + offBf);        // 16*12*512 bf16
    unsigned short* bf2 = bf1 + 16 * 12 * 512;                  // 16*24*512 bf16

    // zero accumulators (H1 + both rowsums contiguous; H2 region of d_out)
    hipMemsetAsync(H1, 0, offRs + (size_t)3 * NN * 4, stream);
    hipMemsetAsync(out_ent, 0, (size_t)NN * DCAT * 4, stream);

    // weight fragments (bf16, WMMA B order) + out_rel
    {
        int tot = 16 * 12 * 512 + 16 * 24 * 512;
        build_bfrags<<<(tot + 255) / 256, 256, 0, stream>>>(a_heads, a_out, bf1, bf2);
    }
    outrel_kernel<<<(NR * DCAT + 255) / 256, 256, 0, stream>>>(rel, W, orel);

    // layer 1: edges -> H1 -> out1 (in place)
    edge_kernel<1><<<2048, 512, 0, stream>>>(ent, rel, el, rt, eln, rtn,
                                             bf1, a2_heads, H1, rs1);
    normalize_kernel<<<(NN * DCAT + 255) / 256, 256, 0, stream>>>(H1, rs1, 2);

    // layer 2: edges -> d_out ent region -> elu(normalized) in place
    edge_kernel<2><<<2048, 512, 0, stream>>>(H1, orel, el, rt, eln, rtn,
                                             bf2, a2_out, out_ent, rs2);
    normalize_kernel<<<(NN * DCAT + 255) / 256, 256, 0, stream>>>(out_ent, rs2, 1);
}
